// LocalLinear_custom_70935679860988
// MI455X (gfx1250) — matrix operand order repaired and checked by reading, not compiled, not run on hardware
//
#include <hip/hip_runtime.h>

typedef __bf16 v8bf  __attribute__((ext_vector_type(8)));
typedef __bf16 v16bf __attribute__((ext_vector_type(16)));
typedef float  v8f   __attribute__((ext_vector_type(8)));

#define K_TOT 576        // C(64) * 3 * 3
#define K_HALF 288       // pairs per row
#define LDS_STRIDE 584   // bf16 elems; 584*2=1168 B: 16B-aligned rows, banks spread (36 dw offset, order 16 mod 64)

// Split fp32 into bf16 hi + bf16 lo (residual), truncation-based (lo absorbs the error).
__device__ __forceinline__ void bf16_split(float v, unsigned short& h, unsigned short& l) {
    unsigned int u = __float_as_uint(v);
    unsigned short hb = (unsigned short)(u >> 16);
    float hf = __uint_as_float((unsigned int)hb << 16);
    float lo = v - hf;
    l = (unsigned short)(__float_as_uint(lo) >> 16);
    h = hb;
}

__device__ __forceinline__ v8bf ld8(const unsigned short* p) {
    return *(const v8bf*)p;   // 16 B aligned by construction -> ds_load_b128
}
#define CAT16(a, b) __builtin_shufflevector(a, b, 0,1,2,3,4,5,6,7,8,9,10,11,12,13,14,15)

__global__ __launch_bounds__(512) void local_linear_wmma_bf16x3(
    const float* __restrict__ x,       // (64, 64, 32, 32)
    const float* __restrict__ weight,  // (64, 32, 32, 64, 3, 3)
    const float* __restrict__ bias,    // (64, 32, 32)
    float* __restrict__ out)           // (64, 64, 32, 32)
{
    // 299 KB static LDS — needs CDNA5's 320 KB/WGP.
    __shared__ unsigned short Ah[64 * LDS_STRIDE];  // patches hi
    __shared__ unsigned short Al[64 * LDS_STRIDE];  // patches lo
    __shared__ unsigned short Bh[64 * LDS_STRIDE];  // weight  hi
    __shared__ unsigned short Bl[64 * LDS_STRIDE];  // weight  lo

    const int pos = blockIdx.x;        // i*32 + j
    const int i   = pos >> 5;
    const int j   = pos & 31;
    const int tid = threadIdx.x;

    // ---- stage + split weight[o, i, j, :] -> Bh/Bl[o][k], float2 loads, u32 packed stores ----
    const float* Wp = weight + (size_t)pos * K_TOT;
    for (int t = tid; t < 64 * K_HALF; t += 512) {
        int o = t / K_HALF;
        int k = (t - o * K_HALF) * 2;
        const float* src = Wp + (size_t)o * (1024u * K_TOT) + k;
        float v0 = src[0], v1 = src[1];
        unsigned short h0, l0, h1, l1;
        bf16_split(v0, h0, l0);
        bf16_split(v1, h1, l1);
        int idx = o * LDS_STRIDE + k;   // k even -> 4B aligned
        *(unsigned int*)&Bh[idx] = (unsigned int)h0 | ((unsigned int)h1 << 16);
        *(unsigned int*)&Bl[idx] = (unsigned int)l0 | ((unsigned int)l1 << 16);
    }

    // ---- stage + split patches[b, i, j, :] -> Ah/Al[b][k] (zero-pad borders) ----
    for (int t = tid; t < 64 * K_HALF; t += 512) {
        int b = t / K_HALF;
        int k = (t - b * K_HALF) * 2;
        unsigned int packed_h = 0, packed_l = 0;
        #pragma unroll
        for (int e = 0; e < 2; ++e) {
            int ke = k + e;
            int c  = ke / 9;
            int r9 = ke - c * 9;
            int kk = r9 / 3;
            int ll = r9 - kk * 3;
            int ih = i + kk - 1;
            int jw = j + ll - 1;
            float v = 0.0f;
            if ((unsigned)ih < 32u && (unsigned)jw < 32u)
                v = x[(((b * 64 + c) * 32 + ih) * 32) + jw];
            unsigned short h, l;
            bf16_split(v, h, l);
            packed_h |= (unsigned int)h << (16 * e);
            packed_l |= (unsigned int)l << (16 * e);
        }
        int idx = b * LDS_STRIDE + k;
        *(unsigned int*)&Ah[idx] = packed_h;
        *(unsigned int*)&Al[idx] = packed_l;
    }
    __syncthreads();

    // ---- 16 waves, one 16x16 tile each; K consumed 32 at a time via bf16 WMMA ----
    const int wave = tid >> 5;
    const int lane = tid & 31;
    const int g    = lane >> 4;
    const int r    = lane & 15;
    const int mt   = wave >> 2;
    const int nt   = wave & 3;

    const unsigned short* Ah_row = &Ah[(mt * 16 + r) * LDS_STRIDE];
    const unsigned short* Al_row = &Al[(mt * 16 + r) * LDS_STRIDE];
    const unsigned short* Bh_row = &Bh[(nt * 16 + r) * LDS_STRIDE];
    const unsigned short* Bl_row = &Bl[(nt * 16 + r) * LDS_STRIDE];

    const float bv = bias[(nt * 16 + r) * 1024 + pos];
    v8f acc = {bv, bv, bv, bv, bv, bv, bv, bv};

    #pragma unroll 3
    for (int k0 = 0; k0 < K_TOT; k0 += 32) {
        // A lane layout (16-bit, 16x32): K chunks at [8g, +8) and [16+8g, +8)
        v16bf ah = CAT16(ld8(Ah_row + k0 + 8 * g), ld8(Ah_row + k0 + 16 + 8 * g));
        v16bf al = CAT16(ld8(Al_row + k0 + 8 * g), ld8(Al_row + k0 + 16 + 8 * g));
        // B lane layout (16-bit, 32x16): K chunks at [8g, +8) and [16+8g, +8)
        v16bf bh = CAT16(ld8(Bh_row + k0 + 8 * g), ld8(Bh_row + k0 + 16 + 8 * g));
        v16bf bl = CAT16(ld8(Bl_row + k0 + 8 * g), ld8(Bl_row + k0 + 16 + 8 * g));
        // fp32 emulation: a*b ~= al*bh + ah*bl + ah*bh (fp32 accumulate in WMMA)
        acc = __builtin_amdgcn_wmma_f32_16x16x32_bf16(false, al, false, bh, (short)0, acc, false, false);
        acc = __builtin_amdgcn_wmma_f32_16x16x32_bf16(false, ah, false, bl, (short)0, acc, false, false);
        acc = __builtin_amdgcn_wmma_f32_16x16x32_bf16(false, ah, false, bh, (short)0, acc, false, false);
    }

    // ---- store: D VGPR vi holds (M = vi + 8*g, N = r) ----
    const int o = nt * 16 + r;
    #pragma unroll
    for (int vi = 0; vi < 8; ++vi) {
        int b = mt * 16 + vi + 8 * g;
        out[((size_t)b * 64 + o) * 1024 + pos] = acc[vi];
    }
}

extern "C" void kernel_launch(void* const* d_in, const int* in_sizes, int n_in,
                              void* d_out, int out_size, void* d_ws, size_t ws_size,
                              hipStream_t stream) {
    const float* x      = (const float*)d_in[0];  // 64*64*32*32
    const float* weight = (const float*)d_in[1];  // 64*32*32*64*3*3
    const float* bias   = (const float*)d_in[2];  // 64*32*32
    float* out = (float*)d_out;                   // 64*64*32*32

    dim3 grid(1024);   // one workgroup per spatial position
    dim3 block(512);   // 16 waves of 32
    local_linear_wmma_bf16x3<<<grid, block, 0, stream>>>(x, weight, bias, out);
}
